// MoeLayer_73839077753277
// MI455X (gfx1250) — matrix operand order repaired and checked by reading, not compiled, not run on hardware
//
#include <hip/hip_runtime.h>

// ---------------- static problem configuration (from reference) ----------------
#define E_    8      // experts
#define G_    8      // groups
#define S_    1024   // tokens per group
#define D_    768    // model dim
#define F_    3072   // ffn dim
#define C_    320    // expert capacity
#define NTOK_ 8192   // total tokens = G_*S_
#define MT    32     // FFN M tile (rows of C)
#define FC    64     // FFN F chunk

typedef __attribute__((ext_vector_type(16))) __bf16 v16bf;
typedef __attribute__((ext_vector_type(8)))  float  v8f;
typedef __attribute__((ext_vector_type(4)))  int    v4i;

union FragU { v16bf v; uint4 q[2]; };

__device__ __forceinline__ unsigned short f2bf(float f) {
  union { float f; unsigned u; } c; c.f = f;
  unsigned r = c.u + 0x7FFFu + ((c.u >> 16) & 1u);   // round-to-nearest-even
  return (unsigned short)(r >> 16);
}

// ---------------- CDNA5 async global->LDS copy path ----------------
#if __has_builtin(__builtin_amdgcn_global_load_async_to_lds_b128) && \
    __has_builtin(__builtin_amdgcn_s_wait_asynccnt)
#define HAS_ASYNC_LDS 1
// one instruction per wave: each lane moves 16B global -> LDS, tracked by ASYNCcnt
__device__ __forceinline__ void async_cp16(unsigned short* lds_dst, const unsigned short* gsrc) {
  __builtin_amdgcn_global_load_async_to_lds_b128((v4i*)(void*)gsrc, (v4i*)(void*)lds_dst, 0, 0);
}
#define WAIT_ASYNC(n) __builtin_amdgcn_s_wait_asynccnt(n)
#else
#define HAS_ASYNC_LDS 0
__device__ __forceinline__ void async_cp16(unsigned short* lds_dst, const unsigned short* gsrc) {
  *(uint4*)lds_dst = *(const uint4*)gsrc;   // synchronous fallback
}
#define WAIT_ASYNC(n) ((void)0)
#endif

// ---------------- 1) router: logits -> softmax -> top-2 ----------------
__global__ __launch_bounds__(256) void topk_kernel(const float* __restrict__ x,
                                                   const float* __restrict__ wr,
                                                   int* __restrict__ topExp,
                                                   float* __restrict__ topGate) {
  __shared__ float wsh[D_ * E_];                       // 24 KB router weights
  const int tid = threadIdx.x;
  for (int i = tid; i < D_ * E_; i += 256) wsh[i] = wr[i];
  __syncthreads();

  const int token = blockIdx.x * 256 + tid;
  const float* xr = x + (size_t)token * D_;
  float acc[E_];
#pragma unroll
  for (int e = 0; e < E_; ++e) acc[e] = 0.f;
  for (int d = 0; d < D_; ++d) {
    float xv = xr[d];
#pragma unroll
    for (int e = 0; e < E_; ++e) acc[e] += xv * wsh[d * E_ + e];
  }
  float m = acc[0];
#pragma unroll
  for (int e = 1; e < E_; ++e) m = fmaxf(m, acc[e]);
  float pr[E_]; float sum = 0.f;
#pragma unroll
  for (int e = 0; e < E_; ++e) { pr[e] = __expf(acc[e] - m); sum += pr[e]; }
  float inv = 1.f / sum;
#pragma unroll
  for (int e = 0; e < E_; ++e) pr[e] *= inv;

  int i1 = 0; float p1 = pr[0];
#pragma unroll
  for (int e = 1; e < E_; ++e) if (pr[e] > p1) { p1 = pr[e]; i1 = e; }
  int i2 = 0; float p2 = -1.f;
#pragma unroll
  for (int e = 0; e < E_; ++e) if (e != i1 && pr[e] > p2) { p2 = pr[e]; i2 = e; }

  topExp[token * 2 + 0] = i1; topGate[token * 2 + 0] = p1;
  topExp[token * 2 + 1] = i2; topGate[token * 2 + 1] = p2;
}

// ---------------- 2) capacity slot assignment (wave32 ballot prefix scan) -----
__global__ __launch_bounds__(32) void assign_kernel(const int* __restrict__ topExp,
                                                    int* __restrict__ routeSlot) {
  const int g = blockIdx.x >> 3;
  const int e = blockIdx.x & 7;
  const int lane = threadIdx.x;          // 0..31, wave32
  int base = 0;
  for (int k = 0; k < 2; ++k) {
    for (int s0 = 0; s0 < S_; s0 += 32) {
      const int s = s0 + lane;
      const int idx = (g * S_ + s) * 2 + k;
      const bool sel = (topExp[idx] == e);
      const unsigned mm = (unsigned)__ballot(sel);
      const int pre = __popc(mm & ((1u << lane) - 1u));
      if (sel) {
        int slot = base + pre;
        routeSlot[idx] = (slot < C_) ? slot : -1;
      }
      base += __popc(mm);
    }
  }
}

// ---------------- 3) weight transpose + f32->bf16 ----------------
// src: [E][R][Cc] f32  ->  dst: [E][Cc][R] bf16
__global__ __launch_bounds__(256) void transpose_bf16_kernel(const float* __restrict__ src,
                                                             unsigned short* __restrict__ dst,
                                                             int R, int Cc) {
  __shared__ unsigned short tile[32][33];
  const int e  = blockIdx.z;
  const int c0 = blockIdx.x * 32;
  const int r0 = blockIdx.y * 32;
  const int tx = threadIdx.x, ty = threadIdx.y;
#pragma unroll
  for (int i = 0; i < 4; ++i) {
    int row = r0 + ty + 8 * i;
    tile[ty + 8 * i][tx] = f2bf(src[((size_t)e * R + row) * Cc + c0 + tx]);
  }
  __syncthreads();
#pragma unroll
  for (int i = 0; i < 4; ++i) {
    int c = c0 + ty + 8 * i;
    dst[((size_t)e * Cc + c) * R + r0 + tx] = tile[tx][ty + 8 * i];
  }
}

// ---------------- 4) dispatch: gather tokens into expert buffers (bf16) ------
__global__ __launch_bounds__(256) void dispatch_kernel(const float* __restrict__ x,
                                                       const int* __restrict__ topExp,
                                                       const int* __restrict__ routeSlot,
                                                       unsigned short* __restrict__ expertIn) {
  const int pair = blockIdx.x;
  const int slot = routeSlot[pair];
  if (slot < 0) return;
  const int token = pair >> 1;
  const int e = topExp[pair];
  const int g = token / S_;
  const float* src = x + (size_t)token * D_;
  unsigned short* dst = expertIn + (((size_t)e * G_ + g) * C_ + slot) * D_;
  for (int d = threadIdx.x; d < D_; d += 256) dst[d] = f2bf(src[d]);
}

// ---------------- 5) fused expert FFN: out = relu(A @ wi) @ wo ----------------
// 256 threads (8 waves) per block, one 32(M) x 768(D) f32 output tile per (e,g).
// K-slabs of 64 double-buffered via async global->LDS; wo^T [768x64] prefetched
// across the whole GEMM1 phase; GEMM2 runs stage-free from LDS.
// Dynamic LDS layout (bf16 elems):
//   As[2][32][64]   (8 KB)
//   Ws[2][64][64]   (16 KB)
//   Hs[32][64]      (4 KB)
//   Wo[768][64]     (96 KB)    total ~124 KB  (<= 320 KB WGP LDS)
#define AS_OFF  0
#define WS_OFF  (2 * MT * 64)
#define HS_OFF  (WS_OFF + 2 * FC * 64)
#define WO_OFF  (HS_OFF + MT * FC)
#define SMEM_ELEMS (WO_OFF + D_ * 64)

__global__ __launch_bounds__(256) void ffn_kernel(const unsigned short* __restrict__ expertIn,
                                                  const unsigned short* __restrict__ wiT,
                                                  const unsigned short* __restrict__ woT,
                                                  float* __restrict__ expertOut) {
  extern __shared__ unsigned short smem[];
  unsigned short* As = smem + AS_OFF;   // [b][r][c] -> (b*MT + r)*64 + c
  unsigned short* Ws = smem + WS_OFF;   // [b][r][c] -> (b*FC + r)*64 + c
  unsigned short* Hs = smem + HS_OFF;   // [r][c]    -> r*64 + c
  unsigned short* Wo = smem + WO_OFF;   // [r][c]    -> r*64 + c

  const int eg = blockIdx.y;
  const int e  = eg >> 3;                 // eg / G_
  const int m0 = blockIdx.x * MT;
  const int tid  = threadIdx.x;
  const int wave = tid >> 5;
  const int lane = tid & 31;
  const int half = lane >> 4;             // lane group 0-15 / 16-31
  const int lm   = lane & 15;
  const int mI = wave & 1;                // 16-row half of the M tile
  const int nQ = wave >> 1;               // 0..3

  const int rB = tid >> 3;                // 0..31  staging row
  const int cO = (tid & 7) * 8;           // 0..56  staging col (8 bf16 = 16B)

  const size_t aBase = ((size_t)eg * C_ + m0) * D_;
  const unsigned short* aSrc  = expertIn + aBase + (size_t)rB * D_ + cO;
  const unsigned short* wSrc0 = wiT + ((size_t)e * F_ + rB) * D_ + cO;
  const unsigned short* wSrc1 = wiT + ((size_t)e * F_ + 32 + rB) * D_ + cO;
  const unsigned short* oSrc  = woT + ((size_t)e * D_ + rB) * F_ + cO;

  v8f acc[12];
#pragma unroll
  for (int j = 0; j < 12; ++j)
#pragma unroll
    for (int r = 0; r < 8; ++r) acc[j][r] = 0.f;

  for (int f0 = 0; f0 < F_; f0 += FC) {
    // prologue: issue K-slab 0 into buffer 0
    async_cp16(&As[rB * 64 + cO], aSrc);
    async_cp16(&Ws[rB * 64 + cO],        wSrc0 + (size_t)f0 * D_);
    async_cp16(&Ws[(32 + rB) * 64 + cO], wSrc1 + (size_t)f0 * D_);

    v8f hacc;
#pragma unroll
    for (int r = 0; r < 8; ++r) hacc[r] = 0.f;

    const int NSLAB = D_ / 64;            // 12
    for (int i = 0; i < NSLAB; ++i) {
      const int k = i * 64;
      if (i + 1 < NSLAB) {                // issue next K-slab (double buffer)
        const int kn = k + 64;
        const int b = (i + 1) & 1;
        async_cp16(&As[(b * MT + rB) * 64 + cO], aSrc + kn);
        async_cp16(&Ws[(b * FC + rB) * 64 + cO],        wSrc0 + (size_t)f0 * D_ + kn);
        async_cp16(&Ws[(b * FC + 32 + rB) * 64 + cO],   wSrc1 + (size_t)f0 * D_ + kn);
      }
      // spread wo^T prefetch (2 rows/thread/stage) across all 12 GEMM1 stages
      async_cp16(&Wo[(i * 64 + rB) * 64 + cO],      oSrc + (size_t)(i * 64) * F_ + f0);
      async_cp16(&Wo[(i * 64 + 32 + rB) * 64 + cO], oSrc + (size_t)(i * 64 + 32) * F_ + f0);

      if (i + 1 < NSLAB) WAIT_ASYNC(5); else WAIT_ASYNC(2);  // slab i landed
      __syncthreads();

      const int b = i & 1;
#pragma unroll
      for (int ks = 0; ks < 64; ks += 32) {
        FragU a, w;
        // A layout: lanes 0-15 hold K0-15, lanes 16-31 K16-31
        a.q[0] = *(const uint4*)&As[(b * MT + 16 * mI + lm) * 64 + ks + 16 * half];
        a.q[1] = *(const uint4*)&As[(b * MT + 16 * mI + lm) * 64 + ks + 16 * half + 8];
        // B layout (B^T rows): lanes 0-15 hold K0-15, lanes 16-31 K16-31
        w.q[0] = *(const uint4*)&Ws[(b * FC + 16 * nQ + lm) * 64 + ks + 16 * half];
        w.q[1] = *(const uint4*)&Ws[(b * FC + 16 * nQ + lm) * 64 + ks + 16 * half + 8];
        hacc = __builtin_amdgcn_wmma_f32_16x16x32_bf16(false, a.v, false, w.v,
                                                       (short)0, hacc, false, false);
      }
      __syncthreads();
    }

    // ReLU -> bf16 -> LDS (C/D layout: vgpr r -> row r + 8*half, col = lane%16)
#pragma unroll
    for (int r = 0; r < 8; ++r)
      Hs[(16 * mI + 8 * half + r) * 64 + 16 * nQ + lm] = f2bf(fmaxf(hacc[r], 0.f));
    WAIT_ASYNC(0);                        // all wo^T parts landed
    __syncthreads();

    // ---- GEMM2: acc[32 x 768] += H[32 x 64] @ wo[64-chunk x 768] ----
#pragma unroll
    for (int k2 = 0; k2 < FC; k2 += 32) {
      FragU a2;
      a2.q[0] = *(const uint4*)&Hs[(16 * mI + lm) * 64 + k2 + 16 * half];
      a2.q[1] = *(const uint4*)&Hs[(16 * mI + lm) * 64 + k2 + 16 * half + 8];
#pragma unroll
      for (int j = 0; j < 12; ++j) {
        FragU b2;
        const int dr = nQ * 192 + j * 16 + lm;
        b2.q[0] = *(const uint4*)&Wo[dr * 64 + k2 + 16 * half];
        b2.q[1] = *(const uint4*)&Wo[dr * 64 + k2 + 16 * half + 8];
        acc[j] = __builtin_amdgcn_wmma_f32_16x16x32_bf16(false, a2.v, false, b2.v,
                                                         (short)0, acc[j], false, false);
      }
    }
    __syncthreads();                      // Wo/Hs reads done before next chunk
  }

#pragma unroll
  for (int j = 0; j < 12; ++j)
#pragma unroll
    for (int r = 0; r < 8; ++r) {
      int row = 16 * mI + 8 * half + r;
      int col = nQ * 192 + j * 16 + lm;
      expertOut[aBase + (size_t)row * D_ + col] = acc[j][r];
    }
}

// ---------------- 6) combine: gated gather back to [B,T,D] ----------------
__global__ __launch_bounds__(256) void combine_kernel(const float* __restrict__ expertOut,
                                                      const int* __restrict__ topExp,
                                                      const int* __restrict__ routeSlot,
                                                      const float* __restrict__ topGate,
                                                      float* __restrict__ out) {
  const int token = blockIdx.x;
  const int g = token / S_;
  const int e0 = topExp[token * 2 + 0], s0 = routeSlot[token * 2 + 0];
  const int e1 = topExp[token * 2 + 1], s1 = routeSlot[token * 2 + 1];
  const float g0 = topGate[token * 2 + 0];
  const float g1 = topGate[token * 2 + 1];
  const float* r0 = expertOut + (((size_t)e0 * G_ + g) * C_ + (s0 < 0 ? 0 : s0)) * D_;
  const float* r1 = expertOut + (((size_t)e1 * G_ + g) * C_ + (s1 < 0 ? 0 : s1)) * D_;
  for (int d = threadIdx.x; d < D_; d += 256) {
    float v = 0.f;
    if (s0 >= 0) v += g0 * r0[d];
    if (s1 >= 0) v += g1 * r1[d];
    out[(size_t)token * D_ + d] = v;
  }
}

// ---------------- host launch ----------------
extern "C" void kernel_launch(void* const* d_in, const int* in_sizes, int n_in,
                              void* d_out, int out_size, void* d_ws, size_t ws_size,
                              hipStream_t stream) {
  (void)in_sizes; (void)n_in; (void)out_size; (void)ws_size;
  const float* x  = (const float*)d_in[0];   // [8,1024,768]
  const float* wr = (const float*)d_in[1];   // [768,8]
  const float* wi = (const float*)d_in[2];   // [8,768,3072]
  const float* wo = (const float*)d_in[3];   // [8,3072,768]
  float* out = (float*)d_out;                // [8,1024,768]

  char* ws = (char*)d_ws; size_t p = 0;
  auto alloc = [&](size_t bytes) -> void* {
    p = (p + 255) & ~(size_t)255;
    void* r = ws + p; p += bytes; return r;
  };
  // ~170 MB total scratch
  unsigned short* expertIn  = (unsigned short*)alloc((size_t)E_ * G_ * C_ * D_ * 2);
  unsigned short* wiT       = (unsigned short*)alloc((size_t)E_ * F_ * D_ * 2);
  unsigned short* woT       = (unsigned short*)alloc((size_t)E_ * D_ * F_ * 2);
  float*          expertOut = (float*)alloc((size_t)E_ * G_ * C_ * D_ * 4);
  int*            routeExp  = (int*)alloc((size_t)NTOK_ * 2 * 4);
  int*            routeSlot = (int*)alloc((size_t)NTOK_ * 2 * 4);
  float*          routeGate = (float*)alloc((size_t)NTOK_ * 2 * 4);

  topk_kernel<<<NTOK_ / 256, 256, 0, stream>>>(x, wr, routeExp, routeGate);
  assign_kernel<<<G_ * E_, 32, 0, stream>>>(routeExp, routeSlot);
  transpose_bf16_kernel<<<dim3(F_ / 32, D_ / 32, E_), dim3(32, 8), 0, stream>>>(wi, wiT, D_, F_);
  transpose_bf16_kernel<<<dim3(D_ / 32, F_ / 32, E_), dim3(32, 8), 0, stream>>>(wo, woT, F_, D_);
  dispatch_kernel<<<NTOK_ * 2, 256, 0, stream>>>(x, routeExp, routeSlot, expertIn);
  ffn_kernel<<<dim3(C_ / MT, E_ * G_), 256, (size_t)SMEM_ELEMS * 2, stream>>>(expertIn, wiT, woT, expertOut);
  combine_kernel<<<NTOK_, 256, 0, stream>>>(expertOut, routeExp, routeSlot, routeGate, out);
}
